// SimpleFineGrainedHead3_63883343561449
// MI455X (gfx1250) — matrix operand order repaired and checked by reading, not compiled, not run on hardware
//
#include <hip/hip_runtime.h>
#include <hip/hip_bf16.h>

// ---------------------------------------------------------------------------
// CDNA5 (gfx1250) implementation: token/similarity GEMMs via v_wmma_f32_16x16x32_f16 (cls GEMM in fp32).
// Fragment layouts per CDNA5 ISA 7.12.2 (16-bit A 16x32, B 32x16, f32 C 16x16).
// ---------------------------------------------------------------------------

typedef __attribute__((ext_vector_type(16))) _Float16 v16h;
typedef __attribute__((ext_vector_type(8)))  float    v8f;

#define B_IMG 96
#define Q_TXT 96
#define V_TOK 197
#define T_TOK 77
#define D_IN  768
#define E_EMB 512
#define SA    520            // padded row stride (halves): 260 dwords == 4 mod 64 -> no LDS bank conflicts
#define QC    8              // q's processed per sim block (A-tile reuse)

union FragU { v16h v; uint4 q[2]; };

// A fragment (16x32, f16) from row-major f16 buffer with row stride `stride` halves.
// Lane (h = lane>>4, r = lane&15): halves 0..7  = A[row0+r][k + h*8 + 0..7]
//                                  halves 8..15 = A[row0+r][k + 16 + h*8 + 0..7]
__device__ __forceinline__ v16h ld_frag_a16(const _Float16* base, int stride, int row0, int k, int lane) {
  const int h = lane >> 4, r = lane & 15;
  const _Float16* p = base + (size_t)(row0 + r) * stride + k + h * 8;
  FragU f;
  f.q[0] = *reinterpret_cast<const uint4*>(p);
  f.q[1] = *reinterpret_cast<const uint4*>(p + 16);
  return f.v;
}

// B fragment (32x16, f16) for D = A * B^T with B rows row-major [N,K]:
// lane (h, r): halves 0..7 = Brow[col0+r][k + h*8 + 0..7], halves 8..15 = Brow[col0+r][k + 16 + h*8 + 0..7]  (N = r, K packed in VGPRs)
__device__ __forceinline__ v16h ld_frag_b16(const _Float16* base, int stride, int col0, int k, int lane) {
  const int h = lane >> 4, r = lane & 15;
  const _Float16* p = base + (size_t)(col0 + r) * stride + k + h * 8;
  FragU f;
  f.q[0] = *reinterpret_cast<const uint4*>(p);
  f.q[1] = *reinterpret_cast<const uint4*>(p + 16);
  return f.v;
}

// A fragment built on the fly from f32 global memory (convert to f16 in registers).
__device__ __forceinline__ v16h ld_frag_a32(const float* X, int stride, int row0, int k, int lane) {
  const int h = lane >> 4, r = lane & 15;
  const float* p = X + (size_t)(row0 + r) * stride + k + h * 8;
  float4 f0 = *reinterpret_cast<const float4*>(p + 0);
  float4 f1 = *reinterpret_cast<const float4*>(p + 4);
  float4 f2 = *reinterpret_cast<const float4*>(p + 16);
  float4 f3 = *reinterpret_cast<const float4*>(p + 20);
  v16h a;
  a[0]  = (_Float16)f0.x; a[1]  = (_Float16)f0.y; a[2]  = (_Float16)f0.z; a[3]  = (_Float16)f0.w;
  a[4]  = (_Float16)f1.x; a[5]  = (_Float16)f1.y; a[6]  = (_Float16)f1.z; a[7]  = (_Float16)f1.w;
  a[8]  = (_Float16)f2.x; a[9]  = (_Float16)f2.y; a[10] = (_Float16)f2.z; a[11] = (_Float16)f2.w;
  a[12] = (_Float16)f3.x; a[13] = (_Float16)f3.y; a[14] = (_Float16)f3.z; a[15] = (_Float16)f3.w;
  return a;
}

// Order-preserving float<->uint encoding so we can use u32 atomicMax as float max.
__device__ __forceinline__ unsigned f2ord(float f) {
  unsigned u = __float_as_uint(f);
  return (u & 0x80000000u) ? ~u : (u | 0x80000000u);
}
__device__ __forceinline__ float ord2f(unsigned u) {
  return (u & 0x80000000u) ? __uint_as_float(u & 0x7fffffffu) : __uint_as_float(~u);
}

// ---------------------------------------------------------------------------
// Kernel 1: f32 -> f16 weight conversion (vectorized).
// ---------------------------------------------------------------------------
__global__ __launch_bounds__(256) void cvt_f32_f16(const float* __restrict__ src,
                                                   _Float16* __restrict__ dst, int n4) {
  int i = blockIdx.x * 256 + threadIdx.x;
  if (i >= n4) return;
  float4 f = reinterpret_cast<const float4*>(src)[i];
  union { _Float16 hh[4]; uint2 u; } o;
  o.hh[0] = (_Float16)f.x; o.hh[1] = (_Float16)f.y;
  o.hh[2] = (_Float16)f.z; o.hh[3] = (_Float16)f.w;
  reinterpret_cast<uint2*>(dst)[i] = o.u;
}

// ---------------------------------------------------------------------------
// Kernel 2: cls projections in full fp32 (tiny GEMM, written straight to d_out).
// ---------------------------------------------------------------------------
__global__ __launch_bounds__(256) void cls_proj(const float* __restrict__ X,
                                                const float* __restrict__ W,
                                                const float* __restrict__ bias,
                                                float* __restrict__ out, int rows) {
  int idx = blockIdx.x * 256 + threadIdx.x;
  if (idx >= rows * E_EMB) return;
  int i = idx >> 9, n = idx & (E_EMB - 1);
  const float4* x = reinterpret_cast<const float4*>(X + (size_t)i * D_IN);
  const float4* w = reinterpret_cast<const float4*>(W + (size_t)n * D_IN);
  float acc = 0.f;
#pragma unroll 4
  for (int t = 0; t < D_IN / 4; ++t) {
    float4 a = x[t], b = w[t];
    acc += a.x * b.x + a.y * b.y + a.z * b.z + a.w * b.w;
  }
  out[idx] = acc + bias[n];
}

// ---------------------------------------------------------------------------
// Kernel 3: token projection + bias + l2norm, output normalized f16 [N,512].
// One block = 16 output rows; 4 waves x 128 cols each (8 WMMA tiles / wave).
// ---------------------------------------------------------------------------
__global__ __launch_bounds__(128) void proj_norm(const float* __restrict__ X,
                                                 const _Float16* __restrict__ Wh,
                                                 const float* __restrict__ bias,
                                                 _Float16* __restrict__ Y) {
  __shared__ float rowss[16];
  __shared__ float ytile[16][SA];   // f32 staging for coalesced f16 stores

  const int tid  = threadIdx.x;
  const int lane = tid & 31;
  const int wv   = tid >> 5;        // 0..3
  const int h    = lane >> 4, r = lane & 15;
  const int row0 = blockIdx.x * 16;

  v8f cc[8];
#pragma unroll
  for (int i = 0; i < 8; ++i)
#pragma unroll
    for (int j = 0; j < 8; ++j) cc[i][j] = 0.f;

  for (int k = 0; k < D_IN; k += 32) {
    v16h a = ld_frag_a32(X, D_IN, row0, k, lane);
#pragma unroll
    for (int nt = 0; nt < 8; ++nt) {
      v16h bb = ld_frag_b16(Wh, D_IN, wv * 128 + nt * 16, k, lane);
      cc[nt] = __builtin_amdgcn_wmma_f32_16x16x32_f16(false, a, false, bb,
                                                      (short)0, cc[nt], false, false);
    }
  }

  if (tid < 16) rowss[tid] = 0.f;
  __syncthreads();

  // bias + per-row sum of squares (C layout: lane holds col n = base+r, rows h*8+j)
  float ssj[8];
#pragma unroll
  for (int j = 0; j < 8; ++j) ssj[j] = 0.f;
#pragma unroll
  for (int nt = 0; nt < 8; ++nt) {
    float bn = bias[wv * 128 + nt * 16 + r];
#pragma unroll
    for (int j = 0; j < 8; ++j) {
      float v = cc[nt][j] + bn;
      cc[nt][j] = v;
      ssj[j] += v * v;
    }
  }
#pragma unroll
  for (int off = 8; off >= 1; off >>= 1)
#pragma unroll
    for (int j = 0; j < 8; ++j) ssj[j] += __shfl_xor(ssj[j], off, 32);
  if (r == 0) {
#pragma unroll
    for (int j = 0; j < 8; ++j) atomicAdd(&rowss[h * 8 + j], ssj[j]);
  }
  __syncthreads();

#pragma unroll
  for (int j = 0; j < 8; ++j) {
    int m = h * 8 + j;
    float sc = 1.f / fmaxf(sqrtf(rowss[m]), 1e-12f);
#pragma unroll
    for (int nt = 0; nt < 8; ++nt)
      ytile[m][wv * 128 + nt * 16 + r] = cc[nt][j] * sc;
  }
  __syncthreads();

  for (int g = tid; g < 16 * (E_EMB / 4); g += 128) {
    int m = g >> 7, c4 = (g & 127) << 2;
    float4 f = *reinterpret_cast<const float4*>(&ytile[m][c4]);
    union { _Float16 hh[4]; uint2 u; } o;
    o.hh[0] = (_Float16)f.x; o.hh[1] = (_Float16)f.y;
    o.hh[2] = (_Float16)f.z; o.hh[3] = (_Float16)f.w;
    *reinterpret_cast<uint2*>(Y + (size_t)(row0 + m) * E_EMB + c4) = o.u;
  }
}

// ---------------------------------------------------------------------------
// Kernel 4: fused similarity + masked max/sum reductions.
// Block = 416 threads (13 waves = 13 v-strips of 16 rows), one b x 8 q's.
// Full v_tok[b] (208x520 f16) + one t_tok[q] (80x520 f16) staged in LDS (~293 KB).
// ---------------------------------------------------------------------------
__global__ __launch_bounds__(416) void sim_kernel(const _Float16* __restrict__ Vt,
                                                  const _Float16* __restrict__ Tt,
                                                  const int* __restrict__ lens,
                                                  float* __restrict__ t2v,
                                                  float* __restrict__ v2t) {
  extern __shared__ char smem_raw[];
  _Float16* Af = reinterpret_cast<_Float16*>(smem_raw);                 // 208 * SA
  _Float16* Bf = reinterpret_cast<_Float16*>(smem_raw) + 208 * SA;      // 80  * SA
  unsigned* colmax = reinterpret_cast<unsigned*>(smem_raw + (size_t)(208 + 80) * SA * 2);
  float* v2t_acc = reinterpret_cast<float*>(colmax + 80);

  const int tid  = threadIdx.x;
  const int lane = tid & 31;
  const int wv   = tid >> 5;        // v-strip 0..12 (rows wv*16 .. wv*16+15)
  const int h    = lane >> 4, r = lane & 15;
  const int b    = blockIdx.y;
  const float NEG = -__builtin_inff();

  // ---- stage A = v_tok[b] (197 rows, zero-pad to 208) ----
  {
    const uint4* src = reinterpret_cast<const uint4*>(Vt + (size_t)b * V_TOK * E_EMB);
    for (int i = tid; i < V_TOK * 64; i += 416) {
      int row = i >> 6, c = i & 63;
      *reinterpret_cast<uint4*>(Af + row * SA + c * 8) = src[row * 64 + c];
    }
    uint4 z = {0u, 0u, 0u, 0u};
    for (int i = tid; i < 11 * 64; i += 416) {
      int row = V_TOK + (i >> 6), c = i & 63;
      *reinterpret_cast<uint4*>(Af + row * SA + c * 8) = z;
    }
  }

  for (int qi = 0; qi < QC; ++qi) {
    const int q = blockIdx.x * QC + qi;
    // ---- stage B = t_tok[q] (77 rows, zero-pad to 80) + init reductions ----
    {
      const uint4* src = reinterpret_cast<const uint4*>(Tt + (size_t)q * T_TOK * E_EMB);
      for (int i = tid; i < T_TOK * 64; i += 416) {
        int row = i >> 6, c = i & 63;
        *reinterpret_cast<uint4*>(Bf + row * SA + c * 8) = src[row * 64 + c];
      }
      uint4 z = {0u, 0u, 0u, 0u};
      for (int i = tid; i < 3 * 64; i += 416) {
        int row = T_TOK + (i >> 6), c = i & 63;
        *reinterpret_cast<uint4*>(Bf + row * SA + c * 8) = z;
      }
      if (tid < 80) colmax[tid] = f2ord(NEG);
      if (tid == 0) *v2t_acc = 0.f;
    }
    __syncthreads();

    const int len = lens[q];

    // ---- 13x5 tile GEMM over K=512, f32 accumulate ----
    v8f cc[5];
#pragma unroll
    for (int t = 0; t < 5; ++t)
#pragma unroll
      for (int j = 0; j < 8; ++j) cc[t][j] = 0.f;

    for (int k = 0; k < E_EMB; k += 32) {
      v16h a = ld_frag_a16(Af, SA, wv * 16, k, lane);
#pragma unroll
      for (int ti = 0; ti < 5; ++ti) {
        v16h bb = ld_frag_b16(Bf, SA, ti * 16, k, lane);
        cc[ti] = __builtin_amdgcn_wmma_f32_16x16x32_f16(false, a, false, bb,
                                                        (short)0, cc[ti], false, false);
      }
    }

    // ---- v2t: per-row max over valid t (0-floored iff len<77), summed over rows ----
    float rm[8];
#pragma unroll
    for (int j = 0; j < 8; ++j) rm[j] = NEG;
#pragma unroll
    for (int ti = 0; ti < 5; ++ti) {
      if (ti * 16 + r < len) {
#pragma unroll
        for (int j = 0; j < 8; ++j) rm[j] = fmaxf(rm[j], cc[ti][j]);
      }
    }
#pragma unroll
    for (int off = 8; off >= 1; off >>= 1)
#pragma unroll
      for (int j = 0; j < 8; ++j) rm[j] = fmaxf(rm[j], __shfl_xor(rm[j], off, 32));
    const float floorv = (len < T_TOK) ? 0.0f : NEG;   // masked zeros enter the t-max
    if (r == 0) {
      float s = 0.f;
#pragma unroll
      for (int j = 0; j < 8; ++j) {
        int m = wv * 16 + h * 8 + j;
        if (m < V_TOK) s += fmaxf(rm[j], floorv);
      }
      atomicAdd(v2t_acc, s);
    }

    // ---- t2v: per-column (t) max over real v rows ----
#pragma unroll
    for (int ti = 0; ti < 5; ++ti) {
      float cm = NEG;
#pragma unroll
      for (int j = 0; j < 8; ++j) {
        int m = wv * 16 + h * 8 + j;
        if (m < V_TOK) cm = fmaxf(cm, cc[ti][j]);
      }
      atomicMax(&colmax[ti * 16 + r], f2ord(cm));
    }
    __syncthreads();

    if (tid == 0) {
      float s = 0.f;
      for (int t = 0; t < len; ++t) s += ord2f(colmax[t]);
      t2v[b * Q_TXT + q] = s / (float)len;          // /= text_sum
      v2t[b * Q_TXT + q] = *v2t_acc / (float)V_TOK; // /= V
    }
    __syncthreads();
  }
}

// ---------------------------------------------------------------------------
// Launcher
// ---------------------------------------------------------------------------
extern "C" void kernel_launch(void* const* d_in, const int* in_sizes, int n_in,
                              void* d_out, int out_size, void* d_ws, size_t ws_size,
                              hipStream_t stream) {
  const float* visual_cls     = (const float*)d_in[0];
  const float* visual_tokens  = (const float*)d_in[1];
  const float* textual_cls    = (const float*)d_in[2];
  const float* textual_tokens = (const float*)d_in[3];
  const float* Wv_cls = (const float*)d_in[4];
  const float* bv_cls = (const float*)d_in[5];
  const float* Wt_cls = (const float*)d_in[6];
  const float* bt_cls = (const float*)d_in[7];
  const float* Wv_tok = (const float*)d_in[8];
  const float* bv_tok = (const float*)d_in[9];
  const float* Wt_tok = (const float*)d_in[10];
  const float* bt_tok = (const float*)d_in[11];
  const int* text_length = (const int*)d_in[12];
  float* out = (float*)d_out;

  // workspace carve-up
  char* ws = (char*)d_ws;
  const size_t wbytes = (size_t)E_EMB * D_IN * 2;                 // 786432
  _Float16* Wv16 = (_Float16*)(ws);
  _Float16* Wt16 = (_Float16*)(ws + wbytes);
  _Float16* Vtn  = (_Float16*)(ws + 2 * wbytes);                  // [18912,512] f16 normalized
  _Float16* Ttn  = (_Float16*)(ws + 2 * wbytes + (size_t)B_IMG * V_TOK * E_EMB * 2);

  // 1) weight conversion
  {
    int n4 = E_EMB * D_IN / 4;
    cvt_f32_f16<<<(n4 + 255) / 256, 256, 0, stream>>>(Wv_tok, Wv16, n4);
    cvt_f32_f16<<<(n4 + 255) / 256, 256, 0, stream>>>(Wt_tok, Wt16, n4);
  }

  // 2) cls projections (fp32, direct to output)
  {
    int n = B_IMG * E_EMB;
    cls_proj<<<(n + 255) / 256, 256, 0, stream>>>(visual_cls, Wv_cls, bv_cls, out, B_IMG);
    cls_proj<<<(n + 255) / 256, 256, 0, stream>>>(textual_cls, Wt_cls, bt_cls,
                                                  out + B_IMG * E_EMB, Q_TXT);
  }

  // 3) token projections + l2norm (WMMA)
  proj_norm<<<(B_IMG * V_TOK) / 16, 128, 0, stream>>>(visual_tokens, Wv16, bv_tok, Vtn);
  proj_norm<<<(Q_TXT * T_TOK) / 16, 128, 0, stream>>>(textual_tokens, Wt16, bt_tok, Ttn);

  // 4) fused similarity + reductions (WMMA, big LDS tiles)
  {
    size_t simLds = (size_t)(208 + 80) * SA * 2 + 80 * 4 + 16;    // ~293 KB < 320 KB/WGP
    (void)hipFuncSetAttribute((const void*)sim_kernel,
                              hipFuncAttributeMaxDynamicSharedMemorySize, (int)simLds);
    float* t2v = out + 2 * B_IMG * E_EMB;
    float* v2t = t2v + B_IMG * Q_TXT;
    sim_kernel<<<dim3(Q_TXT / QC, B_IMG), 416, simLds, stream>>>(Vtn, Ttn, text_length,
                                                                 t2v, v2t);
  }
  (void)in_sizes; (void)n_in; (void)out_size; (void)ws_size;
}
